// MemoryEfficientRGBDecoderStem_90263032693091
// MI455X (gfx1250) — matrix operand order repaired and checked by reading, not compiled, not run on hardware
//
#include <hip/hip_runtime.h>
#include <hip/hip_bf16.h>
#include <math.h>

// ---------------- problem constants (from reference setup) ----------------
#define BATCH  16
#define GDIM   256
#define HIMG   128
#define WIMG   128
#define NF     16
#define KIN    1156          // (256 + 2*16 + 1) * 2 * 2
#define K1PAD  1184          // padded to multiple of 32 (37 k-steps)
#define HID    1024
#define OUTC   256
#define NROWS  65536         // 16 * 64 * 64

typedef __attribute__((ext_vector_type(16))) __bf16 v16bf;
typedef __attribute__((ext_vector_type(8)))  float  v8f;

// ---------------- workspace layout (bytes) ----------------
#define OFF_P   ((size_t)0)
#define OFF_W1  (OFF_P  + (size_t)NROWS * K1PAD * 2)
#define OFF_W2  (OFF_W1 + (size_t)HID   * K1PAD * 2)
#define OFF_H   (OFF_W2 + (size_t)OUTC  * HID   * 2)

// low 32 bits of a flat pointer to __shared__ == wave-relative LDS byte
// address (LDS aperture is selected purely by ADDR[63:32], ISA 10.2)
__device__ __forceinline__ unsigned lds_addr32(const void* p) {
    return (unsigned)(unsigned long long)p;
}

// async copy 16B (global -> LDS), tracked by ASYNCcnt
__device__ __forceinline__ void async_copy16(unsigned lds_off, const void* gaddr) {
    asm volatile("global_load_async_to_lds_b128 %0, %1, off"
                 :: "v"(lds_off), "v"(gaddr) : "memory");
}

// =====================================================================
// Kernel 1: build patchified bf16 input matrix P [NROWS][K1PAD]
// row = b*4096 + py*64 + px ; feature k = c*4 + (dy*2+dx)
// =====================================================================
__global__ void __launch_bounds__(256)
build_p_kernel(const float* __restrict__ pos,
               const float* __restrict__ gestalt,
               const float* __restrict__ mask,
               const float* __restrict__ depth,
               __bf16* __restrict__ P)
{
    int tid = blockIdx.x * blockDim.x + threadIdx.x;   // NROWS*4 threads
    if (tid >= NROWS * 4) return;
    int row = tid >> 2;
    int q   = tid & 3;          // dy*2 + dx
    int b   = row >> 12;
    int py  = (row >> 6) & 63;
    int px  = row & 63;
    int y   = py * 2 + (q >> 1);
    int x   = px * 2 + (q & 1);

    float m = mask [(b * HIMG + y) * WIMG + x];
    float d = depth[(b * HIMG + y) * WIMG + x];

    float posx = fminf(fmaxf(pos[b * 4 + 0], -1.f), 1.f);
    float posy = fminf(fmaxf(pos[b * 4 + 1], -1.f), 1.f);
    float pw   = fminf(fmaxf(pos[b * 4 + 3], 1.f / 128.f), 0.5f);
    float stdv = 0.1f / pw;                       // std_x == std_y (W==H)

    const float HALF_PI = 1.57079632679489662f;
    float gx = (float)x * (2.f / 127.f) - 1.f;
    float gy = (float)y * (2.f / 127.f) - 1.f;
    float ngx = (gx - posx) * stdv * HALF_PI;
    float ngy = (gy - posy) * stdv * HALF_PI;

    __bf16* pr = P + (size_t)row * K1PAD;
    const float* g = gestalt + b * GDIM;

    #pragma unroll 8
    for (int c = 0; c < GDIM; ++c)
        pr[c * 4 + q] = (__bf16)(g[c] * m);

    float fx = ngx, fy = ngy;
    #pragma unroll
    for (int f = 0; f < NF; ++f) {
        pr[(GDIM + 2 * f    ) * 4 + q] = (__bf16)(__cosf(fx) * m);
        pr[(GDIM + 2 * f + 1) * 4 + q] = (__bf16)(__cosf(fy) * m);
        fx *= 2.f; fy *= 2.f;
    }
    pr[(GDIM + 2 * NF) * 4 + q] = (__bf16)(d * m);

    // zero padding K = 1156..1183 (28 elements, 7 per q)
    #pragma unroll
    for (int j = 0; j < 7; ++j)
        pr[KIN + q * 7 + j] = (__bf16)0.f;
}

// =====================================================================
// Weight converters
// =====================================================================
__global__ void __launch_bounds__(256)
cvt_w1_kernel(const float* __restrict__ w1, __bf16* __restrict__ W1b)
{
    int i = blockIdx.x * blockDim.x + threadIdx.x;   // HID*K1PAD
    if (i >= HID * K1PAD) return;
    int r = i / K1PAD, c = i - r * K1PAD;
    float v = (c < KIN) ? w1[(size_t)r * KIN + c] : 0.f;
    W1b[i] = (__bf16)v;
}

__global__ void __launch_bounds__(256)
cvt_w2_kernel(const float* __restrict__ w2, __bf16* __restrict__ W2b)
{
    int i = blockIdx.x * blockDim.x + threadIdx.x;   // OUTC*HID
    if (i >= OUTC * HID) return;
    W2b[i] = (__bf16)w2[i];
}

// =====================================================================
// Tiled WMMA GEMM with double-buffered async global->LDS tile loads:
//   C[M][N] = A[M][K] * W[N][K]^T + bias
// BM=128, BN=128, BK=32. 256 threads = 8 waves (4 along M x 2 along N);
// each wave owns a 32x64 region = 2x4 tiles of 16x16 (v8f accum each).
// Per stage each wave issues exactly 2 async-to-LDS ops, so with one
// stage in flight we wait for ASYNCcnt<=2 before consuming stage k.
// SILU=true : out = silu(v) as bf16 row-major [M][NTOT]   (GEMM1 -> h)
// SILU=false: out = v as f32 scattered to (B, OUTC, 64, 64)  (GEMM2)
// =====================================================================
template<int KDIM, bool SILU>
__global__ void __launch_bounds__(256)
gemm_wmma_kernel(const __bf16* __restrict__ A,
                 const __bf16* __restrict__ Wm,     // [Ncols][KDIM]
                 const float*  __restrict__ bias,
                 __bf16* __restrict__ Hout,         // SILU path
                 float*  __restrict__ Oout,         // !SILU path
                 int NTOT)
{
    __shared__ __align__(16) __bf16 lsa[2][128 * 32];
    __shared__ __align__(16) __bf16 lsb[2][128 * 32];

    const int tid  = threadIdx.x;
    const int lane = tid & 31;
    const int wave = tid >> 5;
    const int wm   = wave & 3;      // 0..3 (M direction, 32 rows each)
    const int wn   = wave >> 2;     // 0..1 (N direction, 64 cols each)
    const int m0   = blockIdx.y * 128;
    const int n0   = blockIdx.x * 128;

    const int half = lane >> 4;     // lane half select
    const int l16  = lane & 15;

    v8f acc[2][4];
    #pragma unroll
    for (int mi = 0; mi < 2; ++mi)
        #pragma unroll
        for (int ni = 0; ni < 4; ++ni)
            acc[mi][ni] = (v8f){0.f, 0.f, 0.f, 0.f, 0.f, 0.f, 0.f, 0.f};

    // cooperative tile loads: thread t covers row (t>>1), 16 cols at (t&1)*16
    const int lr = tid >> 1;
    const int lc = (tid & 1) * 16;

    const unsigned la_off[2] = { lds_addr32(&lsa[0][lr * 32 + lc]),
                                 lds_addr32(&lsa[1][lr * 32 + lc]) };
    const unsigned lb_off[2] = { lds_addr32(&lsb[0][lr * 32 + lc]),
                                 lds_addr32(&lsb[1][lr * 32 + lc]) };
    const __bf16* gA = A  + (size_t)(m0 + lr) * KDIM + lc;
    const __bf16* gB = Wm + (size_t)(n0 + lr) * KDIM + lc;

    constexpr int NKSTEP = KDIM / 32;

    // prologue: stage 0 into buffer 0
    async_copy16(la_off[0], gA);
    async_copy16(lb_off[0], gB);

    for (int k = 0; k < NKSTEP; ++k) {
        const int cur = k & 1;
        if (k + 1 < NKSTEP) {
            // issue next stage into the other buffer
            async_copy16(la_off[cur ^ 1], gA + (k + 1) * 32);
            async_copy16(lb_off[cur ^ 1], gB + (k + 1) * 32);
            asm volatile("s_wait_asynccnt 0x2" ::: "memory");
        } else {
            asm volatile("s_wait_asynccnt 0x0" ::: "memory");
        }
        __syncthreads();   // all waves have stage k resident in LDS

        // A fragment (16x32 bf16): lanes 0-15 hold K {0..7,16..23} of row l16,
        // lanes 16-31 hold K {8..15,24..31}  -> two 16B LDS reads
        v16bf afr[2], bfr[4];
        #pragma unroll
        for (int mi = 0; mi < 2; ++mi) {
            const __bf16* base = &lsa[cur][(wm * 32 + mi * 16 + l16) * 32 + half * 8];
            union { uint4 u[2]; v16bf v; } t;
            t.u[0] = *(const uint4*)(base);
            t.u[1] = *(const uint4*)(base + 16);
            afr[mi] = t.v;
        }
        // B fragment (32x16 bf16): lane holds col N=l16, K {0..7,16..23} (lanes 0-15)
        // or K {8..15,24..31} (lanes 16-31) -> two 16B LDS reads (weight rows are [N][K])
        #pragma unroll
        for (int ni = 0; ni < 4; ++ni) {
            const __bf16* base = &lsb[cur][(wn * 64 + ni * 16 + l16) * 32 + half * 8];
            union { uint4 u[2]; v16bf v; } t;
            t.u[0] = *(const uint4*)(base);
            t.u[1] = *(const uint4*)(base + 16);
            bfr[ni] = t.v;
        }

        #pragma unroll
        for (int mi = 0; mi < 2; ++mi)
            #pragma unroll
            for (int ni = 0; ni < 4; ++ni)
                acc[mi][ni] = __builtin_amdgcn_wmma_f32_16x16x32_bf16(
                    false, afr[mi], false, bfr[ni],
                    (short)0, acc[mi][ni], false, false);

        __syncthreads();   // done reading buf cur before stage k+2 overwrites it
    }

    // ---------------- epilogue ----------------
    #pragma unroll
    for (int mi = 0; mi < 2; ++mi) {
        #pragma unroll
        for (int ni = 0; ni < 4; ++ni) {
            const int ncol = n0 + wn * 64 + ni * 16 + l16;
            const float bv = bias[ncol];
            #pragma unroll
            for (int i = 0; i < 8; ++i) {
                // C/D layout: VGPR i, lanes0-15 -> M=i, lanes16-31 -> M=8+i
                const int mrow = m0 + wm * 32 + mi * 16 + half * 8 + i;
                float v = acc[mi][ni][i] + bv;
                if (SILU) {
                    // silu(v) = v * sigmoid(v); fast rcp instead of IEEE divide
                    v = v * __builtin_amdgcn_rcpf(1.f + __expf(-v));
                    Hout[(size_t)mrow * NTOT + ncol] = (__bf16)v;
                } else {
                    const int b  = mrow >> 12;          // batch
                    const int pp = mrow & 4095;         // py*64 + px
                    Oout[((size_t)b * OUTC + ncol) * 4096 + pp] = v;
                }
            }
        }
    }
}

// =====================================================================
extern "C" void kernel_launch(void* const* d_in, const int* in_sizes, int n_in,
                              void* d_out, int out_size, void* d_ws, size_t ws_size,
                              hipStream_t stream)
{
    const float* pos     = (const float*)d_in[0];
    const float* gestalt = (const float*)d_in[1];
    const float* mask    = (const float*)d_in[2];
    const float* depth   = (const float*)d_in[3];
    const float* weight1 = (const float*)d_in[4];
    const float* bias1   = (const float*)d_in[5];
    const float* weight2 = (const float*)d_in[6];
    const float* bias2   = (const float*)d_in[7];
    float* out = (float*)d_out;

    char* ws = (char*)d_ws;
    __bf16* P    = (__bf16*)(ws + OFF_P);
    __bf16* W1b  = (__bf16*)(ws + OFF_W1);
    __bf16* W2b  = (__bf16*)(ws + OFF_W2);
    __bf16* Hbuf = (__bf16*)(ws + OFF_H);

    // 1) build bf16 patchified input matrix
    build_p_kernel<<<(NROWS * 4) / 256, 256, 0, stream>>>(pos, gestalt, mask, depth, P);

    // 2) convert weights to bf16 (w1 padded along K)
    cvt_w1_kernel<<<(HID * K1PAD + 255) / 256, 256, 0, stream>>>(weight1, W1b);
    cvt_w2_kernel<<<(OUTC * HID + 255) / 256, 256, 0, stream>>>(weight2, W2b);

    // 3) GEMM1 + SiLU -> h (bf16)  [65536 x 1024]
    gemm_wmma_kernel<K1PAD, true><<<dim3(HID / 128, NROWS / 128), 256, 0, stream>>>(
        P, W1b, bias1, Hbuf, nullptr, HID);

    // 4) GEMM2 -> output f32 scattered to (B, 256, 64, 64)
    gemm_wmma_kernel<HID, false><<<dim3(OUTC / 128, NROWS / 128), 256, 0, stream>>>(
        Hbuf, W2b, bias2, nullptr, out, OUTC);
}
